// Encoder_9320079032957
// MI455X (gfx1250) — matrix operand order repaired and checked by reading, not compiled, not run on hardware
//
#include <hip/hip_runtime.h>

// CDNA5 (gfx1250) persistent 2-layer GRU.
//  - rows (B*N = 20800) tiled 32 per block, 650 blocks, hidden state lives in LDS (bf16)
//  - GEMMs via v_wmma_f32_16x16x32_bf16; weights pre-converted to bf16 [N][K=256] in d_ws
//  - weight (B-matrix) tiles loaded with volatile inline-asm global_load_b128 so the
//    compiler cannot hoist them out of the time loop and spill them to scratch
//    (single-buffered: double-buffered pipelining caused accumulator spills)
//  - D=2 input projection of layer0 done with VALU FMAs (too small for WMMA)

typedef __attribute__((ext_vector_type(16))) __bf16 bf16x16;
typedef __attribute__((ext_vector_type(8)))  float  f32x8;
typedef __attribute__((ext_vector_type(4)))  unsigned int u32x4;

union V16 { u32x4 q[2]; bf16x16 v; };

#define WMMA_BF16(a, b, c) \
    __builtin_amdgcn_wmma_f32_16x16x32_bf16(false, (a), false, (b), (short)0, (c), false, false)

// ---- A tile (16x32 bf16) from LDS, row-major stride 256 (compiler ds_load_b128) ----
// lane<16 : row=lane,    K = kb+0..15
// lane>=16: row=lane-16, K = kb+16..31
__device__ __forceinline__ bf16x16 load_a(const __bf16* buf, int rowbase, int kb,
                                          int r, int half) {
    const __bf16* p = buf + (rowbase + r) * 256 + kb + half * 16;
    V16 t;
    t.q[0] = *(const u32x4*)(p);        // 8 bf16
    t.q[1] = *(const u32x4*)(p + 8);    // 8 bf16
    return t.v;
}

// ---- B tiles (32x16 bf16 each) from global bf16 weights stored transposed [N][K=256].
// lane<16 : col=lane, K = kb+0..15 ; lane>=16: col=lane-16, K = kb+16..31
// p points at (cbase*16 + lane&15)*256 + kb + (lane>>4)*16.
// Column-tile stride = 16*256*2 = 8192 bytes; second K-half = +16 bytes.
// volatile asm: issued fresh every k-step, never hoisted/spilled. The trailing
// s_wait_loadcnt overlaps the ds_load_b128 A-tile traffic the compiler schedules
// ahead of this block, and multi-wave occupancy hides the rest of the L2 latency.
__device__ __forceinline__ void load_b4(const __bf16* p, V16 b[4]) {
    asm volatile(
        "global_load_b128 %0, %8, off\n\t"
        "global_load_b128 %1, %8, off offset:16\n\t"
        "global_load_b128 %2, %8, off offset:8192\n\t"
        "global_load_b128 %3, %8, off offset:8208\n\t"
        "global_load_b128 %4, %8, off offset:16384\n\t"
        "global_load_b128 %5, %8, off offset:16400\n\t"
        "global_load_b128 %6, %8, off offset:24576\n\t"
        "global_load_b128 %7, %8, off offset:24592\n\t"
        "s_wait_loadcnt 0x0"
        : "=&v"(b[0].q[0]), "=&v"(b[0].q[1]),
          "=&v"(b[1].q[0]), "=&v"(b[1].q[1]),
          "=&v"(b[2].q[0]), "=&v"(b[2].q[1]),
          "=&v"(b[3].q[0]), "=&v"(b[3].q[1])
        : "v"(p));
}

__device__ __forceinline__ void load_b2(const __bf16* p, V16 b[2]) {
    asm volatile(
        "global_load_b128 %0, %4, off\n\t"
        "global_load_b128 %1, %4, off offset:16\n\t"
        "global_load_b128 %2, %4, off offset:8192\n\t"
        "global_load_b128 %3, %4, off offset:8208\n\t"
        "s_wait_loadcnt 0x0"
        : "=&v"(b[0].q[0]), "=&v"(b[0].q[1]),
          "=&v"(b[1].q[0]), "=&v"(b[1].q[1])
        : "v"(p));
}

__device__ __forceinline__ float sigmoid_f(float v) {
    return 1.0f / (1.0f + __expf(-v));
}
__device__ __forceinline__ float tanh_f(float v) {
    float e = __expf(2.0f * v);
    return 1.0f - 2.0f / (e + 1.0f);
}

// ---------- pass A: zr = sigmoid(u@Wxzr + h@Whzr + bx[:512] + bhzr) ----------
// waves 0..3 produce z (cols 0..255) -> zb ; waves 4..7 produce r*h -> rh
__device__ __forceinline__ void pass_zr(
    const __bf16* u, const __bf16* h,
    const __bf16* Wxzr, const __bf16* Whzr,
    const float* __restrict__ bx, const float* __restrict__ bhzr,
    const float* xb, const float* __restrict__ Wx0,
    __bf16* zb, __bf16* rh, int wave, int r, int half)
{
    f32x8 acc[2][4];
    const int cbase = wave * 4;
    const int boff  = (cbase * 16 + r) * 256 + half * 16;  // element offset of this lane's B base
    for (int c = 0; c < 4; ++c) {
        int col = (cbase + c) * 16 + r;
        float bias = bx[col] + bhzr[col];
        for (int rt = 0; rt < 2; ++rt)
            for (int e = 0; e < 8; ++e) acc[rt][c][e] = bias;
    }
#pragma unroll
    for (int kt = 0; kt < 8; ++kt) {
        int kb = kt * 32;
        bf16x16 a0 = load_a(h, 0, kb, r, half);
        bf16x16 a1 = load_a(h, 16, kb, r, half);
        V16 b[4];
        load_b4(Whzr + boff + kb, b);
#pragma unroll
        for (int c = 0; c < 4; ++c) {
            acc[0][c] = WMMA_BF16(a0, b[c].v, acc[0][c]);
            acc[1][c] = WMMA_BF16(a1, b[c].v, acc[1][c]);
        }
    }
    if (u) {  // layer 1: x-projection is a real GEMM on u = h0
#pragma unroll
        for (int kt = 0; kt < 8; ++kt) {
            int kb = kt * 32;
            bf16x16 a0 = load_a(u, 0, kb, r, half);
            bf16x16 a1 = load_a(u, 16, kb, r, half);
            V16 b[4];
            load_b4(Wxzr + boff + kb, b);
#pragma unroll
            for (int c = 0; c < 4; ++c) {
                acc[0][c] = WMMA_BF16(a0, b[c].v, acc[0][c]);
                acc[1][c] = WMMA_BF16(a1, b[c].v, acc[1][c]);
            }
        }
    }
    for (int c = 0; c < 4; ++c) {
        int col = (cbase + c) * 16 + r;
        float xw0 = 0.f, xw1 = 0.f;
        if (Wx0) { xw0 = Wx0[col]; xw1 = Wx0[768 + col]; }
        for (int rt = 0; rt < 2; ++rt)
            for (int e = 0; e < 8; ++e) {
                int m = rt * 16 + half * 8 + e;
                float v = acc[rt][c][e];
                if (Wx0) v += xb[m * 2] * xw0 + xb[m * 2 + 1] * xw1;
                float s = sigmoid_f(v);
                if (col < 256) {
                    zb[m * 256 + col] = (__bf16)s;
                } else {
                    int hc = col - 256;
                    float hv = (float)h[m * 256 + hc];
                    rh[m * 256 + hc] = (__bf16)(s * hv);
                }
            }
    }
}

// ---------- pass B: h_t = tanh(u@Wxc + rh@Whh + bx[512:] + bhh); h = h + z*(h_t - h)
__device__ __forceinline__ void pass_cand(
    const __bf16* u, __bf16* h,
    const __bf16* Wxc, const __bf16* Whh,
    const float* __restrict__ bx, const float* __restrict__ bhh,
    const float* xb, const float* __restrict__ Wx0,
    const __bf16* zb, const __bf16* rh,
    int wave, int r, int half, float* outp)
{
    f32x8 acc[2][2];
    const int cbase = wave * 2;
    const int boff  = (cbase * 16 + r) * 256 + half * 16;
    for (int c = 0; c < 2; ++c) {
        int col = (cbase + c) * 16 + r;
        float bias = bx[512 + col] + bhh[col];
        for (int rt = 0; rt < 2; ++rt)
            for (int e = 0; e < 8; ++e) acc[rt][c][e] = bias;
    }
#pragma unroll
    for (int kt = 0; kt < 8; ++kt) {
        int kb = kt * 32;
        bf16x16 a0 = load_a(rh, 0, kb, r, half);
        bf16x16 a1 = load_a(rh, 16, kb, r, half);
        V16 b[2];
        load_b2(Whh + boff + kb, b);
#pragma unroll
        for (int c = 0; c < 2; ++c) {
            acc[0][c] = WMMA_BF16(a0, b[c].v, acc[0][c]);
            acc[1][c] = WMMA_BF16(a1, b[c].v, acc[1][c]);
        }
    }
    if (u) {
#pragma unroll
        for (int kt = 0; kt < 8; ++kt) {
            int kb = kt * 32;
            bf16x16 a0 = load_a(u, 0, kb, r, half);
            bf16x16 a1 = load_a(u, 16, kb, r, half);
            V16 b[2];
            load_b2(Wxc + boff + kb, b);
#pragma unroll
            for (int c = 0; c < 2; ++c) {
                acc[0][c] = WMMA_BF16(a0, b[c].v, acc[0][c]);
                acc[1][c] = WMMA_BF16(a1, b[c].v, acc[1][c]);
            }
        }
    }
    for (int c = 0; c < 2; ++c) {
        int col = (cbase + c) * 16 + r;
        float xw0 = 0.f, xw1 = 0.f;
        if (Wx0) { xw0 = Wx0[512 + col]; xw1 = Wx0[768 + 512 + col]; }
        for (int rt = 0; rt < 2; ++rt)
            for (int e = 0; e < 8; ++e) {
                int m = rt * 16 + half * 8 + e;
                float v = acc[rt][c][e];
                if (Wx0) v += xb[m * 2] * xw0 + xb[m * 2 + 1] * xw1;
                float ht = tanh_f(v);
                float z  = (float)zb[m * 256 + col];
                float hv = (float)h[m * 256 + col];
                float hn = hv + z * (ht - hv);
                h[m * 256 + col] = (__bf16)hn;
                if (outp) outp[m * 256 + col] = hn;
            }
    }
}

// ---------- weight prep: fp32 -> bf16, transposed to [N][K=256] ----------
// ws layout (bf16 elems): W0zr_t[512x256] | W0h_t[256x256] | W1xzr_t[512x256] |
//                         W1xc_t[256x256] | W1zr_t[512x256] | W1h_t[256x256]
__global__ void prep_weights(const float* __restrict__ Whzr0, const float* __restrict__ Whh0,
                             const float* __restrict__ Wx1,   const float* __restrict__ Whzr1,
                             const float* __restrict__ Whh1,  __bf16* __restrict__ ws)
{
    int i = blockIdx.x * 256 + threadIdx.x;
    if (i >= 589824) return;
    float v;
    if (i < 131072)      { int j = i;          v = Whzr0[(j & 255) * 512 + (j >> 8)]; }
    else if (i < 196608) { int j = i - 131072; v = Whh0 [(j & 255) * 256 + (j >> 8)]; }
    else if (i < 327680) { int j = i - 196608; v = Wx1  [(j & 255) * 768 + (j >> 8)]; }
    else if (i < 393216) { int j = i - 327680; v = Wx1  [(j & 255) * 768 + 512 + (j >> 8)]; }
    else if (i < 524288) { int j = i - 393216; v = Whzr1[(j & 255) * 512 + (j >> 8)]; }
    else                 { int j = i - 524288; v = Whh1 [(j & 255) * 256 + (j >> 8)]; }
    ws[i] = (__bf16)v;
}

// ---------- persistent GRU kernel: 32 rows / block, T=12 steps ----------
__global__ __launch_bounds__(256)
void gru_kernel(const float* __restrict__ X,
                const float* __restrict__ Wx0, const float* __restrict__ bx0,
                const float* __restrict__ bhzr0, const float* __restrict__ bhh0,
                const float* __restrict__ bx1, const float* __restrict__ bhzr1,
                const float* __restrict__ bhh1,
                const __bf16* __restrict__ ws, float* __restrict__ out)
{
    extern __shared__ __align__(16) char smem[];
    __bf16* h0 = (__bf16*)smem;          // 32x256 bf16
    __bf16* h1 = h0 + 8192;
    __bf16* zb = h1 + 8192;
    __bf16* rh = zb + 8192;
    float*  xb = (float*)(rh + 8192);    // 32x2 f32

    const __bf16* W0zr  = ws;
    const __bf16* W0h   = ws + 131072;
    const __bf16* W1xzr = ws + 196608;
    const __bf16* W1xc  = ws + 327680;
    const __bf16* W1zr  = ws + 393216;
    const __bf16* W1h   = ws + 524288;

    const int tid  = threadIdx.x;
    const int wave = tid >> 5;
    const int lane = tid & 31;
    const int r    = lane & 15;
    const int half = lane >> 4;
    const int g0   = blockIdx.x * 32;

    // zero h0,h1
    {
        unsigned int* p = (unsigned int*)smem;
        for (int i = tid; i < 8192; i += 256) p[i] = 0u;
    }
    __syncthreads();

#pragma unroll 1
    for (int t = 0; t < 12; ++t) {
        if (tid < 64) {  // stage x_t for this block's 32 rows
            int row = tid >> 1, d = tid & 1;
            int g = g0 + row;
            int b = g / 325;
            int nn = g - b * 325;
            xb[tid] = X[(((b * 12) + t) * 325 + nn) * 2 + d];
        }
        __syncthreads();

        // ---- layer 0 (x-projection via VALU, D=2) ----
        pass_zr(nullptr, h0, nullptr, W0zr, bx0, bhzr0, xb, Wx0, zb, rh, wave, r, half);
        __syncthreads();
        pass_cand(nullptr, h0, nullptr, W0h, bx0, bhh0, xb, Wx0, zb, rh, wave, r, half,
                  (t == 11) ? (out + (size_t)g0 * 256) : nullptr);
        __syncthreads();

        // ---- layer 1 (u = h0) ----
        pass_zr(h0, h1, W1xzr, W1zr, bx1, bhzr1, nullptr, nullptr, zb, rh, wave, r, half);
        __syncthreads();
        pass_cand(h0, h1, W1xc, W1h, bx1, bhh1, nullptr, nullptr, zb, rh, wave, r, half,
                  (t == 11) ? (out + 5324800 + (size_t)g0 * 256) : nullptr);
        __syncthreads();
    }
}

extern "C" void kernel_launch(void* const* d_in, const int* in_sizes, int n_in,
                              void* d_out, int out_size, void* d_ws, size_t ws_size,
                              hipStream_t stream) {
    const float* X     = (const float*)d_in[0];
    const float* Wx0   = (const float*)d_in[1];
    const float* bx0   = (const float*)d_in[2];
    const float* Whzr0 = (const float*)d_in[3];
    const float* bhzr0 = (const float*)d_in[4];
    const float* Whh0  = (const float*)d_in[5];
    const float* bhh0  = (const float*)d_in[6];
    const float* Wx1   = (const float*)d_in[7];
    const float* bx1   = (const float*)d_in[8];
    const float* Whzr1 = (const float*)d_in[9];
    const float* bhzr1 = (const float*)d_in[10];
    const float* Whh1  = (const float*)d_in[11];
    const float* bhh1  = (const float*)d_in[12];

    __bf16* ws = (__bf16*)d_ws;
    prep_weights<<<2304, 256, 0, stream>>>(Whzr0, Whh0, Wx1, Whzr1, Whh1, ws);

    const size_t smem = 4 * 16384 + 256;  // 65792 B: h0,h1,z,r*h (bf16) + x stage
    (void)hipFuncSetAttribute((const void*)gru_kernel,
                              hipFuncAttributeMaxDynamicSharedMemorySize, (int)smem);
    gru_kernel<<<650, 256, smem, stream>>>(X, Wx0, bx0, bhzr0, bhh0,
                                           bx1, bhzr1, bhh1, ws, (float*)d_out);
}
